// Head_73650099192165
// MI455X (gfx1250) — matrix operand order repaired and checked by reading, not compiled, not run on hardware
//
#include <hip/hip_runtime.h>

#define B_ 4
#define S_ 4096
#define E_ 1024
#define H_ 128

typedef __bf16 bf16_t;
typedef __bf16 v16bf __attribute__((ext_vector_type(16)));
typedef __bf16 v8bf  __attribute__((ext_vector_type(8)));
typedef float  v8f   __attribute__((ext_vector_type(8)));
typedef float  v4f   __attribute__((ext_vector_type(4)));

static __device__ __forceinline__ v8f wmma_bf16(v16bf a, v16bf b, v8f c) {
    return __builtin_amdgcn_wmma_f32_16x16x32_bf16(false, a, false, b, (short)0, c, false, false);
}

// ---------------------------------------------------------------------------
// Kernel 1: convert & transpose weights fp32 [E,H] -> bf16 Wt[mat][h][e]
// so the WMMA B-operand (contiguous K per lane) becomes one 32B vector load.
// ---------------------------------------------------------------------------
__global__ void wt_convert(const float* __restrict__ Wq,
                           const float* __restrict__ Wk,
                           const float* __restrict__ Wv,
                           bf16_t* __restrict__ Wt) {
    int idx = blockIdx.x * blockDim.x + threadIdx.x;     // 0 .. 3*H*E-1
    int mat = idx / (H_ * E_);
    int r   = idx % (H_ * E_);
    int h   = r / E_;
    int e   = r % E_;
    const float* W = (mat == 0) ? Wq : (mat == 1 ? Wk : Wv);
    Wt[idx] = (bf16_t)W[(size_t)e * H_ + h];
}

// ---------------------------------------------------------------------------
// Kernel 2: QKV projection. One wave = one (matrix, batch, 16-row seq tile);
// computes all 128 output columns via 8 accumulators, K-loop over E in 32s.
// Q,K stored row-major bf16 [b][s][h]; V stored TRANSPOSED bf16 [b][h][s].
// ---------------------------------------------------------------------------
__global__ __launch_bounds__(256) void qkv_proj(const float* __restrict__ x,
                                                const bf16_t* __restrict__ Wt,
                                                const float* __restrict__ bq,
                                                const float* __restrict__ bk,
                                                const float* __restrict__ bv,
                                                bf16_t* __restrict__ Qb,
                                                bf16_t* __restrict__ Kb,
                                                bf16_t* __restrict__ Vt) {
    int wid  = (blockIdx.x * blockDim.x + threadIdx.x) >> 5;
    int lane = threadIdx.x & 31;
    int mat  = wid >> 10;                 // 0=Q 1=K 2=V   (1024 waves each)
    int rem  = wid & 1023;
    int b    = rem >> 8;                  // batch
    int sbase = (rem & 255) << 4;         // seq tile base
    int lo16 = lane & 15;
    int hi   = lane >> 4;

    const bf16_t* wt   = Wt + (size_t)mat * (H_ * E_);
    const float*  bias = (mat == 0) ? bq : (mat == 1 ? bk : bv);

    // A-operand source row for this lane (contiguous K: k = 16*hi + s)
    const float* xrow = x + ((size_t)b * S_ + sbase + lo16) * E_ + hi * 16;

    v8f acc[8];
#pragma unroll
    for (int t = 0; t < 8; ++t) acc[t] = (v8f){0, 0, 0, 0, 0, 0, 0, 0};

    for (int c = 0; c < E_ / 32; ++c) {
        const float* xp = xrow + c * 32;
        v4f f0 = *(const v4f*)(xp);
        v4f f1 = *(const v4f*)(xp + 4);
        v4f f2 = *(const v4f*)(xp + 8);
        v4f f3 = *(const v4f*)(xp + 12);
        v16bf a;
#pragma unroll
        for (int i = 0; i < 4; ++i) {
            a[i]      = (bf16_t)f0[i];
            a[4 + i]  = (bf16_t)f1[i];
            a[8 + i]  = (bf16_t)f2[i];
            a[12 + i] = (bf16_t)f3[i];
        }
#pragma unroll
        for (int t = 0; t < 8; ++t) {
            const bf16_t* wp = wt + (size_t)(t * 16 + lo16) * E_ + c * 32 + hi * 16;
            v16bf bm = *(const v16bf*)wp;               // contiguous K slice
            acc[t] = wmma_bf16(a, bm, acc[t]);
        }
    }

    // bias + store (C layout: lane holds rows r+8*hi, col = t*16 + lo16)
    if (mat == 2) {
#pragma unroll
        for (int t = 0; t < 8; ++t) {
            int col = t * 16 + lo16;
            float bb = bias[col];
#pragma unroll
            for (int r = 0; r < 8; ++r) {
                int srow = sbase + r + 8 * hi;
                Vt[((size_t)b * H_ + col) * S_ + srow] = (bf16_t)(acc[t][r] + bb);
            }
        }
    } else {
        bf16_t* dst = (mat == 0) ? Qb : Kb;
#pragma unroll
        for (int t = 0; t < 8; ++t) {
            int col = t * 16 + lo16;
            float bb = bias[col];
#pragma unroll
            for (int r = 0; r < 8; ++r) {
                int srow = sbase + r + 8 * hi;
                dst[((size_t)b * S_ + srow) * H_ + col] = (bf16_t)(acc[t][r] + bb);
            }
        }
    }
}

// ---------------------------------------------------------------------------
// Kernel 3: flash attention. One 32-thread block (one wave) per 16-query tile.
// Per 32-key step: 8 WMMAs for Q*K^T, online softmax with 16-lane butterflies,
// LDS bounce to convert P to A-layout bf16, 8 WMMAs for P*V (fp32 accum).
// ---------------------------------------------------------------------------
__global__ __launch_bounds__(32) void flash_attn(const bf16_t* __restrict__ Qb,
                                                 const bf16_t* __restrict__ Kb,
                                                 const bf16_t* __restrict__ Vt,
                                                 float* __restrict__ out) {
    __shared__ alignas(64) bf16_t Pl[16 * 32];

    int wid   = blockIdx.x;
    int b     = wid >> 8;
    int qbase = (wid & 255) << 4;
    int lane  = threadIdx.x;
    int lo16  = lane & 15;
    int hi    = lane >> 4;
    const float scale = 0.08838834764831845f;   // 1/sqrt(128)

    // Preload Q in A-layout for the 4 head-dim chunks of 32
    v16bf qa[4];
    {
        const bf16_t* qrow = Qb + ((size_t)b * S_ + qbase + lo16) * H_ + hi * 16;
#pragma unroll
        for (int c = 0; c < 4; ++c) {
            v8bf qlo = *(const v8bf*)(qrow + c * 32);
            v8bf qhi = *(const v8bf*)(qrow + c * 32 + 8);
#pragma unroll
            for (int i = 0; i < 8; ++i) { qa[c][i] = qlo[i]; qa[c][8 + i] = qhi[i]; }
        }
    }

    float m[8], l[8];
    v8f o[8];
#pragma unroll
    for (int r = 0; r < 8; ++r) { m[r] = -1e30f; l[r] = 0.0f; }
#pragma unroll
    for (int t = 0; t < 8; ++t) o[t] = (v8f){0, 0, 0, 0, 0, 0, 0, 0};

    int npairs = (qbase + 16 + 31) >> 5;
    for (int p = 0; p < npairs; ++p) {
        int kb = p << 5;
        v8f c0 = (v8f){0, 0, 0, 0, 0, 0, 0, 0};
        v8f c1 = (v8f){0, 0, 0, 0, 0, 0, 0, 0};
        const bf16_t* krow0 = Kb + ((size_t)b * S_ + kb + lo16) * H_ + hi * 16;
        const bf16_t* krow1 = krow0 + 16 * H_;
        if (p + 1 < npairs) {
            __builtin_prefetch(krow0 + 32 * H_, 0, 0);
            __builtin_prefetch(krow1 + 32 * H_, 0, 0);
        }
#pragma unroll
        for (int c = 0; c < 4; ++c) {
            v16bf b0 = *(const v16bf*)(krow0 + c * 32);
            c0 = wmma_bf16(qa[c], b0, c0);
            v16bf b1 = *(const v16bf*)(krow1 + c * 32);
            c1 = wmma_bf16(qa[c], b1, c1);
        }

        // Online softmax on C-layout: lane holds rows r+8*hi, key col = lo16
        int key0 = kb + lo16;
#pragma unroll
        for (int r = 0; r < 8; ++r) {
            int qidx = qbase + r + 8 * hi;
            float s0 = (key0      <= qidx) ? c0[r] * scale : -1e30f;
            float s1 = (key0 + 16 <= qidx) ? c1[r] * scale : -1e30f;
            float mx = fmaxf(s0, s1);
#pragma unroll
            for (int off = 8; off; off >>= 1) mx = fmaxf(mx, __shfl_xor(mx, off, 32));
            float mnew  = fmaxf(m[r], mx);
            float alpha = __expf(m[r] - mnew);
            m[r] = mnew;
            float p0 = __expf(s0 - mnew);
            float p1 = __expf(s1 - mnew);
            float sum = p0 + p1;
#pragma unroll
            for (int off = 8; off; off >>= 1) sum += __shfl_xor(sum, off, 32);
            l[r] = l[r] * alpha + sum;
#pragma unroll
            for (int t = 0; t < 8; ++t) o[t][r] *= alpha;
            int M = r + 8 * hi;
            Pl[M * 32 + lo16]      = (bf16_t)p0;
            Pl[M * 32 + lo16 + 16] = (bf16_t)p1;
        }
        __syncthreads();

        // Re-read P in A-layout (16x32 bf16)
        v16bf aP;
        {
            const bf16_t* pl = &Pl[lo16 * 32 + hi * 16];
            v8bf plo = *(const v8bf*)pl;
            v8bf phi = *(const v8bf*)(pl + 8);
#pragma unroll
            for (int i = 0; i < 8; ++i) { aP[i] = plo[i]; aP[8 + i] = phi[i]; }
        }
        __syncthreads();

        // O += P * V  (V transposed: contiguous keys per lane)
#pragma unroll
        for (int t = 0; t < 8; ++t) {
            const bf16_t* vp = Vt + ((size_t)b * H_ + t * 16 + lo16) * S_ + kb + hi * 16;
            v16bf vb = *(const v16bf*)vp;
            o[t] = wmma_bf16(aP, vb, o[t]);
        }
    }

    // Normalize and write fp32 output [b][s][h]
#pragma unroll
    for (int r = 0; r < 8; ++r) {
        float inv = 1.0f / l[r];
        int srow = qbase + r + 8 * hi;
#pragma unroll
        for (int t = 0; t < 8; ++t) {
            out[((size_t)b * S_ + srow) * H_ + t * 16 + lo16] = o[t][r] * inv;
        }
    }
}

// ---------------------------------------------------------------------------
extern "C" void kernel_launch(void* const* d_in, const int* in_sizes, int n_in,
                              void* d_out, int out_size, void* d_ws, size_t ws_size,
                              hipStream_t stream) {
    const float* x  = (const float*)d_in[0];
    const float* Wq = (const float*)d_in[1];
    const float* bq = (const float*)d_in[2];
    const float* Wk = (const float*)d_in[3];
    const float* bk = (const float*)d_in[4];
    const float* Wv = (const float*)d_in[5];
    const float* bv = (const float*)d_in[6];
    float* out = (float*)d_out;

    const size_t tensor_bytes = (size_t)B_ * S_ * H_ * sizeof(bf16_t);   // 4 MB
    char* ws = (char*)d_ws;
    bf16_t* Qb = (bf16_t*)(ws);
    bf16_t* Kb = (bf16_t*)(ws + tensor_bytes);
    bf16_t* Vt = (bf16_t*)(ws + 2 * tensor_bytes);
    bf16_t* Wt = (bf16_t*)(ws + 3 * tensor_bytes);                       // 768 KB

    // 1) weights -> bf16 transposed
    wt_convert<<<(3 * H_ * E_) / 256, 256, 0, stream>>>(Wq, Wk, Wv, Wt);

    // 2) QKV projection: 3 * B * (S/16) = 3072 waves, 8 waves/block
    qkv_proj<<<(3 * B_ * (S_ / 16) * 32) / 256, 256, 0, stream>>>(
        x, Wt, bq, bk, bv, Qb, Kb, Vt);

    // 3) flash attention: one wave per 16-query tile
    flash_attn<<<B_ * (S_ / 16), 32, 0, stream>>>(Qb, Kb, Vt, out);
}
